// TransformerEncoderBlock_19559281066516
// MI455X (gfx1250) — matrix operand order repaired and checked by reading, not compiled, not run on hardware
//
#include <hip/hip_runtime.h>
#include <stdint.h>

typedef __attribute__((ext_vector_type(16))) _Float16 v16h;
typedef __attribute__((ext_vector_type(8)))  float    v8f;
typedef __attribute__((ext_vector_type(4)))  unsigned int v4u;

union FragU {
  v16h h;
  v4u  u[2];
};

__device__ __forceinline__ float blockReduceSum(float v, float* red) {
#pragma unroll
  for (int o = 16; o; o >>= 1) v += __shfl_xor(v, o, 32);
  if ((threadIdx.x & 31) == 0) red[threadIdx.x >> 5] = v;
  __syncthreads();
  float s = 0.f;
#pragma unroll
  for (int i = 0; i < 8; ++i) s += red[i];
  __syncthreads();
  return s;
}

// ---------------- f32 -> f16 convert ----------------
__global__ __launch_bounds__(256) void k_f32_to_f16(const float* __restrict__ in,
                                                    _Float16* __restrict__ out,
                                                    long long n) {
  long long i = (long long)blockIdx.x * 256 + threadIdx.x;
  if (i < n) out[i] = (_Float16)in[i];
}

// LDS tile for B operand: 128 columns x 32 K-halfs, padded row stride of
// 40 halfs (80 B) so the 16-lane column gather hits distinct banks.
#define BROW 40

// Cooperative async copy of one B tile (128 rows x 64 bytes) into LDS.
// 256 threads: thread t covers row j = t>>1, 32-byte segment seg = t&1,
// issued as two GLOBAL_LOAD_ASYNC_TO_LDS_B128 (offset applies to both the
// global and the LDS address per the CDNA5 ISA).
__device__ __forceinline__ void asyncLoadTileB(const _Float16* __restrict__ B,
                                               int bn0, int K, int k,
                                               _Float16* sBuf)
{
  const int j   = threadIdx.x >> 1;
  const int seg = threadIdx.x & 1;
  const _Float16* gp = B + (long long)(bn0 + j) * K + (k + seg * 16);
  const unsigned lds = (unsigned)(uintptr_t)&sBuf[j * BROW + seg * 16];
  asm volatile("global_load_async_to_lds_b128 %0, %1, off"
               :: "v"(lds), "v"(gp) : "memory");
  asm volatile("global_load_async_to_lds_b128 %0, %1, off offset:16"
               :: "v"(lds), "v"(gp) : "memory");
}

// ---------------- generic WMMA GEMM ----------------
// C[z] = A[z][M,K] * B[z][N,K]^T  (+bias) * scale  (optional ReLU)
// Block tile 128x128, 8 waves (4 along M x 2 along N), wave tile 32x64.
// B operand double-buffered in LDS via async global->LDS copies.
// SMODE: 0 = f32 out, 1 = f16 out, 2 = f16 transposed out (C[n, m])
template<bool RELU, int SMODE>
__global__ __launch_bounds__(256) void k_gemm(
    const _Float16* __restrict__ A, const _Float16* __restrict__ B,
    const float* __restrict__ bias, void* __restrict__ C,
    int N, int K, float scale,
    long long sA, long long sB, long long sC,
    int ldT, int rowsT, long long bstrideT)
{
  __shared__ _Float16 sBt[2][128 * BROW];

  const int lane = threadIdx.x & 31;
  const int wave = threadIdx.x >> 5;
  const int bm  = blockIdx.x * 128 + (wave & 3) * 32;
  const int bn0 = blockIdx.y * 128;
  const int bnw = (wave >> 2) * 64;          // wave's N offset inside block
  const int z   = blockIdx.z;
  A += (long long)z * sA;
  B += (long long)z * sB;

  v8f acc[2][4];
#pragma unroll
  for (int mi = 0; mi < 2; ++mi)
#pragma unroll
    for (int ni = 0; ni < 4; ++ni) { v8f zz = {}; acc[mi][ni] = zz; }

  const int rA = lane & 15;             // row (A) / col (B) within 16-tile
  const int kA = (lane >> 4) << 3;      // A lane K offset: 0 or 8
  const int kB = (lane >> 4) << 3;      // B lane K offset: 0 or 8

  asyncLoadTileB(B, bn0, K, 0, &sBt[0][0]);

  int cur = 0;
  for (int k = 0; k < K; k += 32) {
    if (k + 32 < K) {
      asyncLoadTileB(B, bn0, K, k + 32, &sBt[cur ^ 1][0]);
      asm volatile("s_wait_asynccnt 2" ::: "memory");   // tile k landed
    } else {
      asm volatile("s_wait_asynccnt 0" ::: "memory");
    }
    __syncthreads();                                     // visible to all waves

    FragU a[2];
#pragma unroll
    for (int mi = 0; mi < 2; ++mi) {
      const _Float16* ap = A + (long long)(bm + mi * 16 + rA) * K + (k + kA);
      a[mi].u[0] = *(const v4u*)ap;          // K = kA .. kA+7
      a[mi].u[1] = *(const v4u*)(ap + 16);   // K = kA+16 .. kA+23
    }
    const _Float16* bbase = &sBt[cur][0];
#pragma unroll
    for (int ni = 0; ni < 4; ++ni) {
      FragU bf;
      const _Float16* bp = bbase + (bnw + ni * 16 + rA) * BROW + kB;
      bf.u[0] = *(const v4u*)bp;             // ds_load_b128: K = kB .. kB+7
      bf.u[1] = *(const v4u*)(bp + 16);      // K = kB+16 .. kB+23
#pragma unroll
      for (int mi = 0; mi < 2; ++mi)
        acc[mi][ni] = __builtin_amdgcn_wmma_f32_16x16x32_f16(
            false, a[mi].h, false, bf.h, (short)0, acc[mi][ni], false, false);
    }
    __syncthreads();                         // reads done before overwrite
    cur ^= 1;
  }

#pragma unroll
  for (int mi = 0; mi < 2; ++mi)
#pragma unroll
    for (int ni = 0; ni < 4; ++ni) {
      const int n  = bn0 + bnw + ni * 16 + (lane & 15);
      const int m0 = bm + mi * 16 + ((lane >> 4) << 3);
      const float bv = bias ? bias[n] : 0.0f;
      if constexpr (SMODE == 2) {
        union { v4u u; _Float16 hv[8]; } pk;
#pragma unroll
        for (int r = 0; r < 8; ++r) {
          float vv = (acc[mi][ni][r] + bv) * scale;
          if (RELU) vv = fmaxf(vv, 0.0f);
          pk.hv[r] = (_Float16)vv;
        }
        const int bb = m0 / rowsT;
        const int ml = m0 - bb * rowsT;
        _Float16* cp = (_Float16*)C + (long long)bb * bstrideT +
                       (long long)n * ldT + ml;
        *(v4u*)cp = pk.u;
      } else {
        const long long base = (long long)z * sC;
#pragma unroll
        for (int r = 0; r < 8; ++r) {
          float vv = (acc[mi][ni][r] + bv) * scale;
          if (RELU) vv = fmaxf(vv, 0.0f);
          const long long idx = base + (long long)(m0 + r) * N + n;
          if constexpr (SMODE == 1) ((_Float16*)C)[idx] = (_Float16)vv;
          else                      ((float*)C)[idx]    = vv;
        }
      }
    }
}

// ---------------- masked softmax over a 2048-length row ----------------
__global__ __launch_bounds__(256) void k_softmax(
    const float* __restrict__ Sc, const unsigned char* __restrict__ mask,
    _Float16* __restrict__ P, int Sl)
{
  __shared__ float red[8];
  const int row = blockIdx.x;              // global query row (b*S + q)
  const int b   = row / Sl;
  const float* src = Sc + (long long)row * Sl;
  const unsigned char* mk = mask + (long long)b * Sl;
  _Float16* dst = P + (long long)row * Sl;
  const int t = threadIdx.x;

  float vals[8];
  float mx = -3.402823466e38f;
#pragma unroll
  for (int i = 0; i < 8; ++i) {
    const int n = t + 256 * i;
    float v = mk[n] ? -3.402823466e38f : src[n];
    vals[i] = v;
    mx = fmaxf(mx, v);
  }
#pragma unroll
  for (int o = 16; o; o >>= 1) mx = fmaxf(mx, __shfl_xor(mx, o, 32));
  if ((t & 31) == 0) red[t >> 5] = mx;
  __syncthreads();
  float gmx = red[0];
#pragma unroll
  for (int i = 1; i < 8; ++i) gmx = fmaxf(gmx, red[i]);
  __syncthreads();

  float s = 0.f;
#pragma unroll
  for (int i = 0; i < 8; ++i) { float e = __expf(vals[i] - gmx); vals[i] = e; s += e; }
  s = blockReduceSum(s, red);
  const float inv = 1.0f / s;
#pragma unroll
  for (int i = 0; i < 8; ++i) dst[t + 256 * i] = (_Float16)(vals[i] * inv);
}

// ---------------- residual add + LayerNorm (E = 768) ----------------
__global__ __launch_bounds__(256) void k_addln(
    const float* __restrict__ X, const float* __restrict__ R,
    const float* __restrict__ g, const float* __restrict__ b,
    float* __restrict__ y32, _Float16* __restrict__ y16)
{
  __shared__ float red[8];
  const long long row = blockIdx.x;
  const float* xp = X + row * 768;
  const float* rp = R + row * 768;
  const int t = threadIdx.x;

  float v[3];
  float s = 0.f;
#pragma unroll
  for (int i = 0; i < 3; ++i) { const int e = t + 256 * i; v[i] = xp[e] + rp[e]; s += v[i]; }
  s = blockReduceSum(s, red);
  const float mean = s * (1.0f / 768.0f);
  float s2 = 0.f;
#pragma unroll
  for (int i = 0; i < 3; ++i) { const float d = v[i] - mean; s2 += d * d; }
  s2 = blockReduceSum(s2, red);
  const float rstd = rsqrtf(s2 * (1.0f / 768.0f) + 1e-5f);
#pragma unroll
  for (int i = 0; i < 3; ++i) {
    const int e = t + 256 * i;
    const float y = (v[i] - mean) * rstd * g[e] + b[e];
    y32[row * 768 + e] = y;
    if (y16) y16[row * 768 + e] = (_Float16)y;
  }
}

extern "C" void kernel_launch(void* const* d_in, const int* in_sizes, int n_in,
                              void* d_out, int out_size, void* d_ws, size_t ws_size,
                              hipStream_t stream) {
  (void)in_sizes; (void)n_in; (void)out_size; (void)ws_size;
  const int B = 8, S = 2048, E = 768, F = 3072;
  const long long M = (long long)B * S;   // 16384 rows

  const float* x     = (const float*)d_in[0];
  const unsigned char* pm = (const unsigned char*)d_in[1];  // jax bool -> 1 byte
  const float* wq_w  = (const float*)d_in[2];
  const float* wq_b  = (const float*)d_in[3];
  const float* wk_w  = (const float*)d_in[4];
  const float* wk_b  = (const float*)d_in[5];
  const float* wv_w  = (const float*)d_in[6];
  const float* wv_b  = (const float*)d_in[7];
  const float* ln1_g = (const float*)d_in[8];
  const float* ln1_b = (const float*)d_in[9];
  const float* m1_w  = (const float*)d_in[10];
  const float* m1_b  = (const float*)d_in[11];
  const float* m2_w  = (const float*)d_in[12];
  const float* m2_b  = (const float*)d_in[13];
  const float* ln2_g = (const float*)d_in[14];
  const float* ln2_b = (const float*)d_in[15];

  // -------- workspace layout (~372 MB, 256B aligned) --------
  char* ws = (char*)d_ws;
  size_t off = 0;
  auto take = [&](size_t bytes) {
    char* p = ws + off;
    off += (bytes + 255) & ~(size_t)255;
    return p;
  };
  _Float16* xh  = (_Float16*)take((size_t)M * E * 2);
  _Float16* wqh = (_Float16*)take((size_t)E * E * 2);
  _Float16* wkh = (_Float16*)take((size_t)E * E * 2);
  _Float16* wvh = (_Float16*)take((size_t)E * E * 2);
  _Float16* m1h = (_Float16*)take((size_t)F * E * 2);
  _Float16* m2h = (_Float16*)take((size_t)E * F * 2);
  _Float16* qh  = (_Float16*)take((size_t)M * E * 2);
  _Float16* kh  = (_Float16*)take((size_t)M * E * 2);  // contiguous after qh
  float*    x1  = (float*)qh;            // overlay: qh+kh dead after scores GEMM
  _Float16* vTh = (_Float16*)take((size_t)M * E * 2);  // [B][E][S]
  float*    sc  = (float*)take((size_t)B * S * S * 4); // scores f32
  _Float16* hid = (_Float16*)sc;         // overlay: scores dead after softmax
  _Float16* P   = (_Float16*)take((size_t)B * S * S * 2);
  float*    mbf = (float*)P;             // overlay: P dead after h GEMM
  float*    hh  = (float*)take((size_t)M * E * 4);
  _Float16* x1h = (_Float16*)take((size_t)M * E * 2);

  auto cvt = [&](const float* src, _Float16* dst, long long n) {
    k_f32_to_f16<<<dim3((unsigned)((n + 255) / 256)), 256, 0, stream>>>(src, dst, n);
  };
  cvt(x,    xh,  M * E);
  cvt(wq_w, wqh, (long long)E * E);
  cvt(wk_w, wkh, (long long)E * E);
  cvt(wv_w, wvh, (long long)E * E);
  cvt(m1_w, m1h, (long long)F * E);
  cvt(m2_w, m2h, (long long)E * F);

  const float qscale = 0.036084391824351615f;  // 1/sqrt(768), folded into q

  // q = (x Wq^T + bq) * scale  -> f16 [M,E]
  k_gemm<false, 1><<<dim3(128, 6, 1), 256, 0, stream>>>(
      xh, wqh, wq_b, qh, E, E, qscale, 0, 0, 0, 0, 1, 0);
  // k = x Wk^T + bk -> f16 [M,E]
  k_gemm<false, 1><<<dim3(128, 6, 1), 256, 0, stream>>>(
      xh, wkh, wk_b, kh, E, E, 1.0f, 0, 0, 0, 0, 1, 0);
  // v = x Wv^T + bv -> f16 transposed [B][E][S]
  k_gemm<false, 2><<<dim3(128, 6, 1), 256, 0, stream>>>(
      xh, wvh, wv_b, vTh, E, E, 1.0f, 0, 0, 0, S, S, (long long)E * S);

  // scores[b] = q[b] k[b]^T -> f32 [B][S][S]
  k_gemm<false, 0><<<dim3(16, 16, 8), 256, 0, stream>>>(
      qh, kh, nullptr, sc, S, E, 1.0f,
      (long long)S * E, (long long)S * E, (long long)S * S, 0, 1, 0);

  // masked softmax -> P f16 [B][S][S]
  k_softmax<<<dim3((unsigned)M), 256, 0, stream>>>(sc, pm, P, S);

  // h[b] = P[b] v[b] -> f32 [B][S][E]   (B operand = vT, K-contiguous)
  k_gemm<false, 0><<<dim3(16, 6, 8), 256, 0, stream>>>(
      P, vTh, nullptr, hh, E, S, 1.0f,
      (long long)S * S, (long long)E * S, (long long)S * E, 0, 1, 0);

  // x1 = LN(x + h); also f16 copy for MLP
  k_addln<<<dim3((unsigned)M), 256, 0, stream>>>(x, hh, ln1_g, ln1_b, x1, x1h);

  // hidden = relu(x1 W1^T + b1) -> f16 [M,F]
  k_gemm<true, 1><<<dim3(128, 24, 1), 256, 0, stream>>>(
      x1h, m1h, m1_b, hid, F, E, 1.0f, 0, 0, 0, 0, 1, 0);

  // m = relu(hidden W2^T + b2) -> f32 [M,E]
  k_gemm<true, 0><<<dim3(128, 6, 1), 256, 0, stream>>>(
      hid, m2h, m2_b, mbf, E, F, 1.0f, 0, 0, 0, 0, 1, 0);

  // out = LN(x1 + m)
  k_addln<<<dim3((unsigned)M), 256, 0, stream>>>(
      x1, mbf, ln2_g, ln2_b, (float*)d_out, (_Float16*)nullptr);
}
